// SymmetricalResidualGAT_6906307412649
// MI455X (gfx1250) — matrix operand order repaired and checked by reading, not compiled, not run on hardware
//
#include <hip/hip_runtime.h>

// ---------------------------------------------------------------------------
// CDNA5 (gfx1250) wave32 WMMA types
// ---------------------------------------------------------------------------
typedef __attribute__((ext_vector_type(16))) __bf16 v16bf;
typedef __attribute__((ext_vector_type(8)))  float  v8f;

union FragBF { uint4 q[2]; v16bf v; };

__device__ __forceinline__ unsigned short f2bf(float f) {
  unsigned int u = __float_as_uint(f);
  u += 0x7FFFu + ((u >> 16) & 1u);          // round-to-nearest-even
  return (unsigned short)(u >> 16);
}

// monotone float -> u32 mapping for atomicMax on floats
__device__ __forceinline__ unsigned int encf(float f) {
  int i = __float_as_int(f);
  return (unsigned int)(i >= 0 ? (i ^ (int)0x80000000) : ~i);
}
__device__ __forceinline__ float decf(unsigned int u) {
  return (u & 0x80000000u) ? __uint_as_float(u ^ 0x80000000u)
                           : __uint_as_float(~u);
}
#define ENC_NEGINF 0x007FFFFFu
#define NEG_SLOPE 0.2f

__device__ __forceinline__ v8f wmma_bf16(const FragBF& a, const FragBF& b, v8f c) {
  // D = A(16x32 bf16) x B(32x16 bf16) + C(16x16 f32)
  return __builtin_amdgcn_wmma_f32_16x16x32_bf16(false, a.v, false, b.v,
                                                 (short)0, c, false, false);
}

// A frag (16-bit A 16x32): lane = row (0..15), half-wave selects
// K = kk+half*16 .. kk+half*16+15 (contiguous, matches B frag)
__device__ __forceinline__ void load_afrag(FragBF& a, const unsigned short* base,
                                           int rowStride, int lane, int kk) {
  const int half = lane >> 4;
  const uint4* ap = (const uint4*)(base + (lane & 15) * rowStride + kk + half * 16);
  a.q[0] = ap[0];
  a.q[1] = ap[1];
}

// B frag (16-bit B 32x16, same contiguous k order as A frag): lane = col (0..15),
// lanes 0-15 hold K=kk..kk+15, lanes 16-31 hold K=kk+16..kk+31 (contiguous)
__device__ __forceinline__ void load_bfrag(FragBF& b, const unsigned short* Wt,
                                           size_t K, int col, int lane, int kk) {
  const int half = lane >> 4;
  const uint4* bp = (const uint4*)(Wt + (size_t)col * K + kk + half * 16);
  b.q[0] = bp[0];
  b.q[1] = bp[1];
}

// ---------------------------------------------------------------------------
// Utility kernels
// ---------------------------------------------------------------------------
__global__ void fill_u32_kernel(unsigned int* __restrict__ p, unsigned int v, long long n) {
  long long i = (long long)blockIdx.x * blockDim.x + threadIdx.x;
  long long stride = (long long)gridDim.x * blockDim.x;
  for (; i < n; i += stride) p[i] = v;
}

// in-place: ordered-u32 -> float, ENC_NEGINF (no contribution) -> 0.0f
__global__ void decode_max_kernel(unsigned int* __restrict__ p, long long n) {
  long long i = (long long)blockIdx.x * blockDim.x + threadIdx.x;
  long long stride = (long long)gridDim.x * blockDim.x;
  for (; i < n; i += stride) {
    unsigned int u = p[i];
    float f = (u == ENC_NEGINF) ? 0.f : decf(u);
    ((float*)p)[i] = f;
  }
}

// W[K,NOUT] f32 -> Wt[NOUT,K] bf16 (transposed so WMMA B-frags are contiguous)
__global__ void wconvert_kernel(const float* __restrict__ W, unsigned short* __restrict__ Wt,
                                int K, int NOUT) {
  long long n = (long long)K * NOUT;
  long long i = (long long)blockIdx.x * blockDim.x + threadIdx.x;
  long long stride = (long long)gridDim.x * blockDim.x;
  for (; i < n; i += stride) {
    int k = (int)(i / NOUT), c = (int)(i % NOUT);
    Wt[(size_t)c * K + k] = f2bf(W[i]);
  }
}

// ---------------------------------------------------------------------------
// Generic bf16-WMMA GEMM: C[M,NOUT] = A[M,K](f32) * W (Wt = [NOUT,K] bf16) + bias?
// block = 256 thr = 8 waves; tile = 16 rows x 128 cols; grid.y covers NOUT/128.
// Full 16xK A-panel staged once in LDS -> barrier-free WMMA inner loop.
// ---------------------------------------------------------------------------
__global__ __launch_bounds__(256) void gemm_bf16_kernel(
    const float* __restrict__ A, const unsigned short* __restrict__ Wt,
    const float* __restrict__ bias, float* __restrict__ C,
    int M, int K, int NOUT)
{
  __shared__ __align__(16) unsigned short As[16 * 256];    // 8 KB (K <= 256)
  const int t = threadIdx.x;
  const int wave = t >> 5, lane = t & 31;
  const int half = lane >> 4, lcol = lane & 15;
  const int row0 = blockIdx.x * 16;
  const int col0 = blockIdx.y * 128 + wave * 16;

  // stage full A panel (f32 -> bf16), coalesced along K
  for (int j = 0; j < 16; ++j) {
    int gr = row0 + j;
    for (int c = t; c < K; c += 256) {
      float v = (gr < M) ? A[(size_t)gr * K + c] : 0.f;
      As[j * K + c] = f2bf(v);
    }
  }
  __syncthreads();

  v8f acc = {0.f, 0.f, 0.f, 0.f, 0.f, 0.f, 0.f, 0.f};
  for (int kk = 0; kk < K; kk += 32) {
    FragBF a, b;
    load_afrag(a, As, K, lane, kk);
    __builtin_prefetch((const void*)(Wt + (size_t)(col0 + lcol) * K + kk + 64), 0, 0);
    load_bfrag(b, Wt, (size_t)K, col0 + lcol, lane, kk);
    acc = wmma_bf16(a, b, acc);
  }

  float bb = bias ? bias[col0 + lcol] : 0.f;
#pragma unroll
  for (int r = 0; r < 8; ++r) {
    int grow = row0 + half * 8 + r;            // C/D: lane=col, VGPR r -> row r+8*half
    if (grow < M) C[(size_t)grow * NOUT + col0 + lcol] = acc[r] + bb;
  }
}

// ---------------------------------------------------------------------------
// Fused EdgeConv: per-edge MLP relu([xd | xs-xd] W1 + b1) W2 + b2, seg-max @ dst
// block = 16 edges; full K=256 message tile staged in LDS as bf16
// ---------------------------------------------------------------------------
__global__ __launch_bounds__(256) void edgeconv_kernel(
    const float* __restrict__ x, const long long* __restrict__ ei,
    const unsigned short* __restrict__ wt1, const float* __restrict__ b1,
    const unsigned short* __restrict__ wt2, const float* __restrict__ b2,
    unsigned int* __restrict__ agg_enc, int N, long long E)
{
  __shared__ __align__(16) unsigned short Msg[16 * 256];   // 8 KB
  __shared__ __align__(16) unsigned short H1 [16 * 128];   // 4 KB
  __shared__ int SS[16], SD[16];
  const int t = threadIdx.x;
  const long long eb = (long long)blockIdx.x * 16;

  if (t < 16) {
    long long e = eb + t;
    int s = -1, d = -1;
    if (e < E) { s = (int)ei[e]; d = (int)ei[E + e]; }
    SS[t] = s; SD[t] = d;
  }
  __syncthreads();

  for (int j = 0; j < 16; ++j) {               // gather message rows, cvt bf16
    int d = SD[j], s = SS[j];
    float v = 0.f;
    if (d >= 0) {
      if (t < 128) v = x[(size_t)d * 128 + t];
      else { int c = t - 128; v = x[(size_t)s * 128 + c] - x[(size_t)d * 128 + c]; }
    }
    Msg[j * 256 + t] = f2bf(v);
  }
  __syncthreads();

  const int wave = t >> 5, lane = t & 31;
  const int half = lane >> 4, lcol = lane & 15;
  const int col0 = wave * 16;                  // 8 waves x 16 = 128 cols

  // layer 1: [16x256] x [256x128]
  v8f acc = {0.f, 0.f, 0.f, 0.f, 0.f, 0.f, 0.f, 0.f};
  for (int kk = 0; kk < 256; kk += 32) {
    FragBF a, b;
    load_afrag(a, Msg, 256, lane, kk);
    load_bfrag(b, wt1, 256, col0 + lcol, lane, kk);
    acc = wmma_bf16(a, b, acc);
  }
  float bb1 = b1[col0 + lcol];
#pragma unroll
  for (int r = 0; r < 8; ++r) {
    float v = acc[r] + bb1;
    v = v > 0.f ? v : 0.f;                     // relu
    H1[(half * 8 + r) * 128 + col0 + lcol] = f2bf(v);
  }
  __syncthreads();

  // layer 2: [16x128] x [128x128]
  v8f acc2 = {0.f, 0.f, 0.f, 0.f, 0.f, 0.f, 0.f, 0.f};
  for (int kk = 0; kk < 128; kk += 32) {
    FragBF a, b;
    load_afrag(a, H1, 128, lane, kk);
    load_bfrag(b, wt2, 128, col0 + lcol, lane, kk);
    acc2 = wmma_bf16(a, b, acc2);
  }
  float bb2 = b2[col0 + lcol];
#pragma unroll
  for (int r = 0; r < 8; ++r) {
    int row = half * 8 + r;
    int d = SD[row];
    if (d >= 0) {
      float v = acc2[r] + bb2;
      atomicMax(&agg_enc[(size_t)d * 128 + col0 + lcol], encf(v));
    }
  }
}

// ---------------------------------------------------------------------------
// GAT attention kernels (edge passes are gather/atomic bound -> VALU)
// ---------------------------------------------------------------------------
__global__ void att_logits_kernel(const float* __restrict__ h,
                                  const float* __restrict__ a_s, const float* __restrict__ a_d,
                                  float* __restrict__ al_s, float* __restrict__ al_d, int N)
{
  int n = blockIdx.x * blockDim.x + threadIdx.x;
  if (n >= N) return;
#pragma unroll
  for (int hh = 0; hh < 2; ++hh) {
    float s0 = 0.f, s1 = 0.f;
    const float* hp = h + (size_t)n * 256 + hh * 128;
    for (int c = 0; c < 128; ++c) {
      float v = hp[c];
      s0 += v * a_s[hh * 128 + c];
      s1 += v * a_d[hh * 128 + c];
    }
    al_s[n * 2 + hh] = s0;
    al_d[n * 2 + hh] = s1;
  }
}

__global__ void edge_max_kernel(const long long* __restrict__ ei,
                                const float* __restrict__ al_s, const float* __restrict__ al_d,
                                unsigned int* __restrict__ m_enc, int N, long long E)
{
  long long i = (long long)blockIdx.x * blockDim.x + threadIdx.x;
  if (i >= E + N) return;
  int s, d;
  if (i < E) { s = (int)ei[i]; d = (int)ei[E + i]; }
  else       { s = d = (int)(i - E); }          // self loops
#pragma unroll
  for (int hh = 0; hh < 2; ++hh) {
    float e = al_s[s * 2 + hh] + al_d[d * 2 + hh];
    e = e > 0.f ? e : NEG_SLOPE * e;
    atomicMax(&m_enc[d * 2 + hh], encf(e));
  }
}

__global__ void edge_sum_kernel(const long long* __restrict__ ei,
                                const float* __restrict__ al_s, const float* __restrict__ al_d,
                                const float* __restrict__ m, float* __restrict__ ssum,
                                int N, long long E)
{
  long long i = (long long)blockIdx.x * blockDim.x + threadIdx.x;
  if (i >= E + N) return;
  int s, d;
  if (i < E) { s = (int)ei[i]; d = (int)ei[E + i]; }
  else       { s = d = (int)(i - E); }
#pragma unroll
  for (int hh = 0; hh < 2; ++hh) {
    float e = al_s[s * 2 + hh] + al_d[d * 2 + hh];
    e = e > 0.f ? e : NEG_SLOPE * e;
    atomicAdd(&ssum[d * 2 + hh], expf(e - m[d * 2 + hh]));
  }
}

// one wave per edge; 256 features scattered with atomicAdd
__global__ __launch_bounds__(256) void edge_agg_kernel(
    const long long* __restrict__ ei, const float* __restrict__ h,
    const float* __restrict__ al_s, const float* __restrict__ al_d,
    const float* __restrict__ m, const float* __restrict__ ssum,
    float* __restrict__ out, int N, long long E)
{
  long long wid = (long long)blockIdx.x * (blockDim.x >> 5) + (threadIdx.x >> 5);
  int lane = threadIdx.x & 31;
  if (wid >= E + N) return;
  int s, d;
  if (wid < E) { s = (int)ei[wid]; d = (int)ei[E + wid]; }
  else         { s = d = (int)(wid - E); }
  float alpha[2];
#pragma unroll
  for (int hh = 0; hh < 2; ++hh) {
    float e = al_s[s * 2 + hh] + al_d[d * 2 + hh];
    e = e > 0.f ? e : NEG_SLOPE * e;
    float p = expf(e - m[d * 2 + hh]);
    alpha[hh] = p / (ssum[d * 2 + hh] + 1e-16f);
  }
  const float* hs = h + (size_t)s * 256;
  float* od = out + (size_t)d * 256;
#pragma unroll
  for (int rep = 0; rep < 8; ++rep) {
    int c = rep * 32 + lane;
    atomicAdd(&od[c], hs[c] * alpha[c >> 7]);
  }
}

// out = relu(out + bias) [+ residual]
__global__ void epilogue_kernel(float* __restrict__ out, const float* __restrict__ bias,
                                const float* __restrict__ residual, int N)
{
  long long n = (long long)N * 256;
  long long i = (long long)blockIdx.x * blockDim.x + threadIdx.x;
  long long stride = (long long)gridDim.x * blockDim.x;
  for (; i < n; i += stride) {
    float v = out[i] + bias[(int)(i & 255)];
    v = v > 0.f ? v : 0.f;
    if (residual) v += residual[i];
    out[i] = v;
  }
}

// out[n] = x[n,:] . w + b   (one wave per row, shuffle reduction, wave32)
__global__ __launch_bounds__(256) void final_kernel(const float* __restrict__ x,
                                                    const float* __restrict__ w,
                                                    const float* __restrict__ b,
                                                    float* __restrict__ out, int N)
{
  int wid = blockIdx.x * (blockDim.x >> 5) + (threadIdx.x >> 5);
  int lane = threadIdx.x & 31;
  if (wid >= N) return;
  float p = 0.f;
  for (int c = lane; c < 256; c += 32) p += x[(size_t)wid * 256 + c] * w[c];
#pragma unroll
  for (int off = 16; off > 0; off >>= 1) p += __shfl_xor(p, off, 32);
  if (lane == 0) out[wid] = p + b[0];
}

// ---------------------------------------------------------------------------
// Host orchestration
// ---------------------------------------------------------------------------
extern "C" void kernel_launch(void* const* d_in, const int* in_sizes, int n_in,
                              void* d_out, int out_size, void* d_ws, size_t ws_size,
                              hipStream_t stream)
{
  const float*     x     = (const float*)d_in[0];
  const long long* ei    = (const long long*)d_in[1];
  const float*     ec_w1 = (const float*)d_in[3];
  const float*     ec_b1 = (const float*)d_in[4];
  const float*     ec_w2 = (const float*)d_in[5];
  const float*     ec_b2 = (const float*)d_in[6];
  const float*     reg_w = (const float*)d_in[7];
  const float*     reg_b = (const float*)d_in[8];
  const float*     fin_w = (const float*)d_in[9];
  const float*     fin_b = (const float*)d_in[10];
  const float* gw [4] = {(const float*)d_in[11], (const float*)d_in[15],
                         (const float*)d_in[19], (const float*)d_in[23]};
  const float* gas[4] = {(const float*)d_in[12], (const float*)d_in[16],
                         (const float*)d_in[20], (const float*)d_in[24]};
  const float* gad[4] = {(const float*)d_in[13], (const float*)d_in[17],
                         (const float*)d_in[21], (const float*)d_in[25]};
  const float* gb [4] = {(const float*)d_in[14], (const float*)d_in[18],
                         (const float*)d_in[22], (const float*)d_in[26]};

  const int Nn = in_sizes[0] / 128;
  const long long E = in_sizes[1] / 2;
  const long long T = E + Nn;

  // --- workspace carve-up (256B aligned regions) ---
  char* base = (char*)d_ws;
  size_t off = 0;
  auto carve = [&](size_t bytes) -> char* {
    char* p = base + off;
    off = (off + bytes + 255) & ~(size_t)255;
    return p;
  };
  unsigned short* wt_ec1 = (unsigned short*)carve((size_t)128 * 256 * 2);
  unsigned short* wt_ec2 = (unsigned short*)carve((size_t)128 * 128 * 2);
  unsigned short* wt_g[4];
  wt_g[0] = (unsigned short*)carve((size_t)256 * 128 * 2);
  wt_g[1] = (unsigned short*)carve((size_t)256 * 256 * 2);
  wt_g[2] = (unsigned short*)carve((size_t)256 * 256 * 2);
  wt_g[3] = (unsigned short*)carve((size_t)256 * 256 * 2);
  unsigned short* wt_reg = (unsigned short*)carve((size_t)256 * 256 * 2);
  unsigned int* agg_enc  = (unsigned int*)carve((size_t)Nn * 128 * 4);  // -> x0 in place
  float* hbuf = (float*)carve((size_t)Nn * 256 * 4);
  float* bufA = (float*)carve((size_t)Nn * 256 * 4);
  float* bufB = (float*)carve((size_t)Nn * 256 * 4);
  float* al_s = (float*)carve((size_t)Nn * 2 * 4);
  float* al_d = (float*)carve((size_t)Nn * 2 * 4);
  unsigned int* m_enc = (unsigned int*)carve((size_t)Nn * 2 * 4);       // -> m in place
  float* ssum = (float*)carve((size_t)Nn * 2 * 4);
  (void)ws_size; (void)n_in; (void)out_size;

  auto blk = [](long long n, int per) { return (unsigned)((n + per - 1) / per); };

  // --- weights: f32 -> transposed bf16 ---
  wconvert_kernel<<<blk(256 * 128, 256), 256, 0, stream>>>(ec_w1, wt_ec1, 256, 128);
  wconvert_kernel<<<blk(128 * 128, 256), 256, 0, stream>>>(ec_w2, wt_ec2, 128, 128);
  wconvert_kernel<<<blk(128 * 256, 256), 256, 0, stream>>>(gw[0], wt_g[0], 128, 256);
  for (int i = 1; i < 4; ++i)
    wconvert_kernel<<<blk(256 * 256, 256), 256, 0, stream>>>(gw[i], wt_g[i], 256, 256);
  wconvert_kernel<<<blk(256 * 256, 256), 256, 0, stream>>>(reg_w, wt_reg, 256, 256);

  // --- EdgeConv + segment-max -> x0 [N,128] ---
  fill_u32_kernel<<<blk((long long)Nn * 128, 256), 256, 0, stream>>>(agg_enc, ENC_NEGINF, (long long)Nn * 128);
  edgeconv_kernel<<<blk(E, 16), 256, 0, stream>>>(x, ei, wt_ec1, ec_b1, wt_ec2, ec_b2, agg_enc, Nn, E);
  decode_max_kernel<<<blk((long long)Nn * 128, 256), 256, 0, stream>>>(agg_enc, (long long)Nn * 128);
  const float* x0 = (const float*)agg_enc;

  // --- one GATConv layer ---
  auto run_gat = [&](const float* xin, int K, const unsigned short* wt,
                     const float* a_s, const float* a_d, const float* bvec,
                     const float* residual, float* xout) {
    dim3 gg(blk(Nn, 16), 2);
    gemm_bf16_kernel<<<gg, 256, 0, stream>>>(xin, wt, nullptr, hbuf, Nn, K, 256);
    att_logits_kernel<<<blk(Nn, 256), 256, 0, stream>>>(hbuf, a_s, a_d, al_s, al_d, Nn);
    fill_u32_kernel<<<blk((long long)Nn * 2, 256), 256, 0, stream>>>(m_enc, ENC_NEGINF, (long long)Nn * 2);
    fill_u32_kernel<<<blk((long long)Nn * 2, 256), 256, 0, stream>>>((unsigned int*)ssum, 0u, (long long)Nn * 2);
    edge_max_kernel<<<blk(T, 256), 256, 0, stream>>>(ei, al_s, al_d, m_enc, Nn, E);
    decode_max_kernel<<<blk((long long)Nn * 2, 256), 256, 0, stream>>>(m_enc, (long long)Nn * 2);
    edge_sum_kernel<<<blk(T, 256), 256, 0, stream>>>(ei, al_s, al_d, (const float*)m_enc, ssum, Nn, E);
    fill_u32_kernel<<<blk((long long)Nn * 256, 256), 256, 0, stream>>>((unsigned int*)xout, 0u, (long long)Nn * 256);
    edge_agg_kernel<<<blk(T, 8), 256, 0, stream>>>(ei, hbuf, al_s, al_d, (const float*)m_enc, ssum, xout, Nn, E);
    epilogue_kernel<<<blk((long long)Nn * 256, 256), 256, 0, stream>>>(xout, bvec, residual, Nn);
  };

  // g1: x1 = relu(gat(x0))
  run_gat(x0, 128, wt_g[0], gas[0], gad[0], gb[0], nullptr, bufA);
  // g2: x2 = relu(gat(x1)) + x1
  run_gat(bufA, 256, wt_g[1], gas[1], gad[1], gb[1], bufA, bufB);
  // reg: x3 = x2 @ reg_w + reg_b
  {
    dim3 gg(blk(Nn, 16), 2);
    gemm_bf16_kernel<<<gg, 256, 0, stream>>>(bufB, wt_reg, reg_b, bufA, Nn, 256, 256);
  }
  // g3: x4 = relu(gat(x3)) + x3
  run_gat(bufA, 256, wt_g[2], gas[2], gad[2], gb[2], bufA, bufB);
  // g4: x5 = relu(gat(x4))
  run_gat(bufB, 256, wt_g[3], gas[3], gad[3], gb[3], nullptr, bufA);
  // final: out = x5 @ fin_w + fin_b
  final_kernel<<<blk(Nn, 8), 256, 0, stream>>>(bufA, fin_w, fin_b, (float*)d_out, Nn);
}
